// MaskedSelfAttention_8890582303334
// MI455X (gfx1250) — matrix operand order repaired and checked by reading, not compiled, not run on hardware
//
#include <hip/hip_runtime.h>

// MI455X / gfx1250, wave32. Masked causal self-attention, flash-attention style.
// Precision: f16 WMMA (v_wmma_f32_16x16x32_f16) with f32 accumulation.

#define B_ 4
#define S_ 4096
#define D_ 1024
#define H_ 128

typedef __attribute__((ext_vector_type(16))) _Float16 v16h;
typedef __attribute__((ext_vector_type(8)))  float    v8f;

union HFrag {
  v16h v;
  uint4 q[2];
  _Float16 h[16];
};

union H4 {
  _Float16 h[4];
  uint2 u;
};

static __device__ __forceinline__ v8f wmma_f16(v16h a, v16h b, v8f c) {
  return __builtin_amdgcn_wmma_f32_16x16x32_f16(false, a, false, b,
                                                (short)0, c, false, false);
}

// ---------------------------------------------------------------------------
// Kernel 1: QKV projection.  out[s,h] = sum_d x[s,d] * W[h,d] + bias[h]
// Block: 256 threads (8 waves). Block tile: M=64 rows x N=32 cols, K-step 64.
// Q,K stored row-major f16 [B*S][H]; V stored transposed f16 [B][H][S].
// ---------------------------------------------------------------------------
__global__ __launch_bounds__(256)
void qkv_proj_kernel(const float* __restrict__ x,
                     const float* __restrict__ Wq, const float* __restrict__ bq,
                     const float* __restrict__ Wk, const float* __restrict__ bk,
                     const float* __restrict__ Wv, const float* __restrict__ bv,
                     _Float16* __restrict__ Qh, _Float16* __restrict__ Kh,
                     _Float16* __restrict__ Vt)
{
  __shared__ alignas(16) _Float16 xls[64][72];  // [m][k], +8 pad
  __shared__ alignas(16) _Float16 wls[32][72];  // [n][k], +8 pad (col-major B)

  const int tid  = threadIdx.x;
  const int lane = tid & 31;
  const int wave = tid >> 5;
  const int mg   = wave & 3;   // row group 0..3  (16 rows each)
  const int ngc  = wave >> 2;  // col group 0..1  (16 cols each)

  const int ms0 = blockIdx.x * 64;      // global row in [0, B*S)
  const int nt  = blockIdx.y;           // 0..11
  const int mat = nt >> 2;              // 0=Q 1=K 2=V
  const int h0  = (nt & 3) * 32;

  const float* W    = (mat == 0) ? Wq : (mat == 1) ? Wk : Wv;
  const float* bias = (mat == 0) ? bq : (mat == 1) ? bk : bv;

  const int ml    = lane & 15;
  const int hw8   = (lane & 16) ? 8 : 0;    // A-frag K sub-chunk select
  const int khalf = (lane & 16) ? 16 : 0;   // B-frag K half select

  // Per-thread staging coordinates (16 float4 per 64-wide row)
  const int xrow = tid >> 4;            // 0..15 (advance by 16 per step)
  const int xc4  = (tid & 15) * 4;      // 0,4,...,60

  v8f acc = {};

  for (int k0 = 0; k0 < D_; k0 += 64) {
    // stage x tile 64x64 (f32 -> f16): 4 x float4 per thread, unrolled
#pragma unroll
    for (int i = 0; i < 4; ++i) {
      int row = xrow + i * 16;
      float4 f = *(const float4*)&x[(size_t)(ms0 + row) * D_ + k0 + xc4];
      H4 p;
      p.h[0] = (_Float16)f.x; p.h[1] = (_Float16)f.y;
      p.h[2] = (_Float16)f.z; p.h[3] = (_Float16)f.w;
      *(uint2*)&xls[row][xc4] = p.u;
    }
    // stage W tile 32x64 as [n][k]: 2 x float4 per thread
#pragma unroll
    for (int i = 0; i < 2; ++i) {
      int row = xrow + i * 16;
      float4 f = *(const float4*)&W[(size_t)(h0 + row) * D_ + k0 + xc4];
      H4 p;
      p.h[0] = (_Float16)f.x; p.h[1] = (_Float16)f.y;
      p.h[2] = (_Float16)f.z; p.h[3] = (_Float16)f.w;
      *(uint2*)&wls[row][xc4] = p.u;
    }
    __syncthreads();

    HFrag a0, a1, b0, b1;
    a0.q[0] = *(const uint4*)&xls[mg * 16 + ml][hw8];
    a0.q[1] = *(const uint4*)&xls[mg * 16 + ml][16 + hw8];
    a1.q[0] = *(const uint4*)&xls[mg * 16 + ml][32 + hw8];
    a1.q[1] = *(const uint4*)&xls[mg * 16 + ml][48 + hw8];
    b0.q[0] = *(const uint4*)&wls[ngc * 16 + ml][hw8];
    b0.q[1] = *(const uint4*)&wls[ngc * 16 + ml][16 + hw8];
    b1.q[0] = *(const uint4*)&wls[ngc * 16 + ml][32 + hw8];
    b1.q[1] = *(const uint4*)&wls[ngc * 16 + ml][48 + hw8];
    acc = wmma_f16(a0.v, b0.v, acc);
    acc = wmma_f16(a1.v, b1.v, acc);
    __syncthreads();
  }

  // Epilogue: C layout -> element (m = r + 8*(lane>=16), n = lane&15)
  const int ncol = lane & 15;
  const int hi8  = (lane & 16) ? 8 : 0;
  const int hcol = h0 + ngc * 16 + ncol;
  const float bb = bias[hcol];
  const int bidx = ms0 / S_;            // 64 | S_, tile never crosses batch
#pragma unroll
  for (int r = 0; r < 8; ++r) {
    int row = ms0 + mg * 16 + hi8 + r;
    _Float16 val = (_Float16)(acc[r] + bb);
    if (mat == 0)      Qh[(size_t)row * H_ + hcol] = val;
    else if (mat == 1) Kh[(size_t)row * H_ + hcol] = val;
    else {
      int s = row - bidx * S_;
      Vt[((size_t)bidx * H_ + hcol) * S_ + s] = val;  // V transposed [b][h][s]
    }
  }
}

// ---------------------------------------------------------------------------
// Kernel 2: flash attention forward with causal + padding mask.
// Block: (b, 128-query tile), 8 waves; wave w owns query rows [q0+16w, +16).
// Per 32-key tile: scores = Q K^T (8 wmma), online softmax (intra-16-lane
// shuffles), P -> LDS -> A-fragment, out += P V (8 wmma).
// ---------------------------------------------------------------------------
__global__ __launch_bounds__(256)
void attn_kernel(const _Float16* __restrict__ Qh, const _Float16* __restrict__ Kh,
                 const _Float16* __restrict__ Vt, const int* __restrict__ mask,
                 float* __restrict__ out)
{
  __shared__ alignas(16) _Float16 Kls[32][136];     // [key][h], +8 pad
  __shared__ alignas(16) _Float16 Vls[128][40];     // [h][key], +8 pad
  __shared__ alignas(16) _Float16 Pls[8][16][40];   // per-wave P tile [m][kv]
  __shared__ int mls[32];

  const int tid   = threadIdx.x;
  const int lane  = tid & 31;
  const int wave  = tid >> 5;
  const int b     = blockIdx.y;
  const int q0    = blockIdx.x * 128;
  const int qbase = q0 + wave * 16;

  const int ncol  = lane & 15;                 // C-layout col / A-frag row / B-frag col
  const int hi8   = (lane & 16) ? 8 : 0;       // C-layout row offset / A-frag chunk
  const int khalf = (lane & 16) ? 16 : 0;      // B-frag K half

  const float NEG_INF = -__builtin_inff();
  const float sc = 0.08838834764831845f;       // 1/sqrt(128)

  // Per-thread staging coordinates
  const int krow = tid >> 3, kcol = (tid & 7) * 16;   // K tile [32][128]
  const int vrow = tid >> 1, vcol = (tid & 1) * 16;   // V tile [128][32]
  const _Float16* Kbase = Kh + (size_t)b * S_ * H_;
  const _Float16* Vbase = Vt + (size_t)b * H_ * S_;

  // Resident Q A-fragments: 16 rows x 128 (4 K-steps of 32)
  HFrag qa[4];
  {
    const _Float16* qrow = Qh + ((size_t)b * S_ + qbase + ncol) * H_;
#pragma unroll
    for (int ks = 0; ks < 4; ++ks) {
      qa[ks].q[0] = *(const uint4*)(qrow + ks * 32 + hi8);
      qa[ks].q[1] = *(const uint4*)(qrow + ks * 32 + 16 + hi8);
    }
  }

  v8f acc[8];
  {
    v8f z = {};
#pragma unroll
    for (int g = 0; g < 8; ++g) acc[g] = z;
  }
  float run_m[8], run_l[8];
#pragma unroll
  for (int r = 0; r < 8; ++r) { run_m[r] = NEG_INF; run_l[r] = 0.f; }

  const int kv_tiles = (q0 + 128) / 32;
  for (int t = 0; t < kv_tiles; ++t) {
    const int kv0 = t * 32;

    // ---- stage K tile [32][128]: 16 halves (32B) per thread, coalesced ----
    {
      const uint4* src = (const uint4*)(Kbase + (size_t)(kv0 + krow) * H_ + kcol);
      uint4* dst = (uint4*)&Kls[krow][kcol];
      dst[0] = src[0]; dst[1] = src[1];
    }
    // ---- stage V tile [128][32] from Vt[b][h][kv0..] ----
    {
      const uint4* src = (const uint4*)(Vbase + (size_t)vrow * S_ + kv0 + vcol);
      uint4* dst = (uint4*)&Vls[vrow][vcol];
      dst[0] = src[0]; dst[1] = src[1];
    }
    if (tid < 32) mls[tid] = mask[b * S_ + kv0 + tid];

    // ---- prefetch next tile's K/V lines while this tile computes ----
    if (t + 1 < kv_tiles) {
      const int nkv = kv0 + 32;
      __builtin_prefetch(Kbase + (size_t)(nkv + krow) * H_ + kcol, 0, 1);
      __builtin_prefetch(Vbase + (size_t)vrow * S_ + nkv + vcol, 0, 1);
    }
    __syncthreads();

    if (kv0 <= qbase + 15) {   // tile not fully above the diagonal for this wave
      // ---- scores: S = Q K^T, two 16-key column groups ----
      v8f s0 = {}, s1 = {};
#pragma unroll
      for (int ks = 0; ks < 4; ++ks) {
        HFrag bk0, bk1;
        bk0.q[0] = *(const uint4*)&Kls[ncol][ks * 32 + hi8];
        bk0.q[1] = *(const uint4*)&Kls[ncol][ks * 32 + 16 + hi8];
        bk1.q[0] = *(const uint4*)&Kls[16 + ncol][ks * 32 + hi8];
        bk1.q[1] = *(const uint4*)&Kls[16 + ncol][ks * 32 + 16 + hi8];
        s0 = wmma_f16(qa[ks].v, bk0.v, s0);
        s1 = wmma_f16(qa[ks].v, bk1.v, s1);
      }

      // ---- mask + online softmax (rows live across 16-lane halves) ----
      const int msk0 = mls[ncol], msk1 = mls[16 + ncol];
      const int kva = kv0 + ncol, kvb = kv0 + 16 + ncol;
      float f0[8], f1[8], rm[8], alpha[8];
#pragma unroll
      for (int r = 0; r < 8; ++r) {
        int qi = qbase + hi8 + r;
        f0[r] = (msk0 != 0 && kva <= qi) ? s0[r] * sc : NEG_INF;
        f1[r] = (msk1 != 0 && kvb <= qi) ? s1[r] * sc : NEG_INF;
        float v = fmaxf(f0[r], f1[r]);
#pragma unroll
        for (int off = 1; off < 16; off <<= 1)
          v = fmaxf(v, __shfl_xor(v, off, 32));
        rm[r] = v;
      }
#pragma unroll
      for (int r = 0; r < 8; ++r) {
        float nm = fmaxf(run_m[r], rm[r]);
        alpha[r] = (nm == NEG_INF) ? 1.0f : __expf(run_m[r] - nm);
        run_m[r] = nm;
      }
#pragma unroll
      for (int r = 0; r < 8; ++r) {
        float nm = run_m[r];
        float p0 = (nm == NEG_INF) ? 0.f : __expf(f0[r] - nm);
        float p1 = (nm == NEG_INF) ? 0.f : __expf(f1[r] - nm);
        float srow = p0 + p1;
#pragma unroll
        for (int off = 1; off < 16; off <<= 1)
          srow += __shfl_xor(srow, off, 32);
        run_l[r] = run_l[r] * alpha[r] + srow;
        Pls[wave][hi8 + r][ncol]      = (_Float16)p0;  // C-layout -> LDS
        Pls[wave][hi8 + r][16 + ncol] = (_Float16)p1;
      }
#pragma unroll
      for (int g = 0; g < 8; ++g)
#pragma unroll
        for (int r = 0; r < 8; ++r) acc[g][r] *= alpha[r];

      // ---- out += P V : reload P as A-fragment (same wave, dscnt-ordered) ----
      HFrag pa;
      pa.q[0] = *(const uint4*)&Pls[wave][ncol][hi8];
      pa.q[1] = *(const uint4*)&Pls[wave][ncol][16 + hi8];
#pragma unroll
      for (int g = 0; g < 8; ++g) {
        HFrag bv;
        bv.q[0] = *(const uint4*)&Vls[g * 16 + ncol][hi8];
        bv.q[1] = *(const uint4*)&Vls[g * 16 + ncol][16 + hi8];
        acc[g] = wmma_f16(pa.v, bv.v, acc[g]);
      }
    }
    __syncthreads();   // protect K/V LDS before next tile's staging
  }

  // ---- epilogue: normalize and store f32 ----
#pragma unroll
  for (int g = 0; g < 8; ++g) {
#pragma unroll
    for (int r = 0; r < 8; ++r) {
      float l = run_l[r];
      float val = (l > 0.f) ? acc[g][r] / l : 0.f;
      out[((size_t)b * S_ + qbase + hi8 + r) * H_ + g * 16 + ncol] = val;
    }
  }
}

// ---------------------------------------------------------------------------
extern "C" void kernel_launch(void* const* d_in, const int* in_sizes, int n_in,
                              void* d_out, int out_size, void* d_ws, size_t ws_size,
                              hipStream_t stream) {
  const float* x  = (const float*)d_in[0];
  const int* mask = (const int*)d_in[1];
  const float* Wq = (const float*)d_in[2];
  const float* bq = (const float*)d_in[3];
  const float* Wk = (const float*)d_in[4];
  const float* bk = (const float*)d_in[5];
  const float* Wv = (const float*)d_in[6];
  const float* bv = (const float*)d_in[7];
  float* out = (float*)d_out;

  _Float16* Qh = (_Float16*)d_ws;                    // [B*S][H] f16
  _Float16* Kh = Qh + (size_t)B_ * S_ * H_;          // [B*S][H] f16
  _Float16* Vt = Kh + (size_t)B_ * S_ * H_;          // [B][H][S] f16 (transposed)

  qkv_proj_kernel<<<dim3((B_ * S_) / 64, 12), 256, 0, stream>>>(
      x, Wq, bq, Wk, bk, Wv, bv, Qh, Kh, Vt);
  attn_kernel<<<dim3(S_ / 128, B_), 256, 0, stream>>>(Qh, Kh, Vt, mask, out);
}
